// CatRSDNet1_77326591197599
// MI455X (gfx1250) — matrix operand order repaired and checked by reading, not compiled, not run on hardware
//
#include <hip/hip_runtime.h>
#include <hip/hip_bf16.h>

#define T_LEN 8192
#define FEAT  1664
#define HID   128
#define NM    5
#define G4    512            // 4*HID gates per model
#define GTOT  2560           // NM * G4

typedef __bf16 bf16;
typedef bf16  v16bf __attribute__((ext_vector_type(16)));
typedef float v8f   __attribute__((ext_vector_type(8)));
typedef float v2f   __attribute__((ext_vector_type(2)));

union FragU { v16bf v; uint4 q[2]; };

// ---------------------------------------------------------------------------
// Kernel 1: split X (fp32) into bf16 hi/lo pair  (x = hi + lo to ~fp32 acc)
// ---------------------------------------------------------------------------
__global__ void split_x_kernel(const float* __restrict__ X,
                               bf16* __restrict__ Xh, bf16* __restrict__ Xl) {
    int i = blockIdx.x * blockDim.x + threadIdx.x;
    if (i >= T_LEN * FEAT) return;
    float x = X[i];
    bf16 h = (bf16)x;
    Xh[i] = h;
    Xl[i] = (bf16)(x - (float)h);
}

// ---------------------------------------------------------------------------
// Kernel 2: fold mask_x into W_ih and split into bf16 hi/lo
// ---------------------------------------------------------------------------
__global__ void split_w_kernel(const float* __restrict__ W_ih,
                               const float* __restrict__ mask_x,
                               bf16* __restrict__ Wh, bf16* __restrict__ Wl) {
    int i = blockIdx.x * blockDim.x + threadIdx.x;
    if (i >= GTOT * FEAT) return;
    int g = i / FEAT;
    int f = i - g * FEAT;
    int r = g >> 9;                       // 512 gates per model
    float w = W_ih[i] * mask_x[r * FEAT + f];
    bf16 h = (bf16)w;
    Wh[i] = h;
    Wl[i] = (bf16)(w - (float)h);
}

// ---------------------------------------------------------------------------
// Kernel 3: gx = X @ W'^T  via v_wmma_f32_16x16x32_bf16, 3-product split.
// Block = 256 thr (8 waves), WG tile 64(M) x 128(N), wave tile 32x32.
// Fragments fed straight from global: Xh+Xl (54MB) + Wh+Wl (17MB) fit in
// the 192MB L2, so no LDS staging stage is needed.
// ---------------------------------------------------------------------------
__global__ __launch_bounds__(256) void
gemm_gx_kernel(const bf16* __restrict__ Xh, const bf16* __restrict__ Xl,
               const bf16* __restrict__ Wh, const bf16* __restrict__ Wl,
               float* __restrict__ gx) {
    const int lane = threadIdx.x & 31;
    const int wid  = threadIdx.x >> 5;
    const int wm   = wid >> 2;            // 0..1
    const int wn   = wid & 3;             // 0..3
    const int m0   = blockIdx.y * 64 + wm * 32;
    const int n0   = blockIdx.x * 128 + wn * 32;

    v8f acc[2][2];
    #pragma unroll
    for (int i = 0; i < 2; ++i)
        #pragma unroll
        for (int j = 0; j < 2; ++j)
            acc[i][j] = (v8f)(0.f);

    const int l15 = lane & 15;
    const int lhi = lane >> 4;

    for (int k0 = 0; k0 < FEAT; k0 += 32) {
        FragU ah[2], al[2], bh[2], bl[2];
        // A (16x32 bf16): lanes 0-15 row m, K {0..7},{16..23}; lanes 16-31 K {8..15},{24..31}
        const int ka = k0 + (lhi << 3);
        #pragma unroll
        for (int i = 0; i < 2; ++i) {
            size_t o = (size_t)(m0 + i * 16 + l15) * FEAT + ka;
            ah[i].q[0] = *(const uint4*)(Xh + o);
            ah[i].q[1] = *(const uint4*)(Xh + o + 16);
            al[i].q[0] = *(const uint4*)(Xl + o);
            al[i].q[1] = *(const uint4*)(Xl + o + 16);
        }
        // B (32x16 bf16): lanes 0-15 col n, K {0..7},{16..23}; lanes 16-31 K {8..15},{24..31}
        const int kb = k0 + (lhi << 3);
        #pragma unroll
        for (int j = 0; j < 2; ++j) {
            size_t o = (size_t)(n0 + j * 16 + l15) * FEAT + kb;
            bh[j].q[0] = *(const uint4*)(Wh + o);
            bh[j].q[1] = *(const uint4*)(Wh + o + 16);
            bl[j].q[0] = *(const uint4*)(Wl + o);
            bl[j].q[1] = *(const uint4*)(Wl + o + 16);
        }
        #pragma unroll
        for (int i = 0; i < 2; ++i)
            #pragma unroll
            for (int j = 0; j < 2; ++j) {
                acc[i][j] = __builtin_amdgcn_wmma_f32_16x16x32_bf16(
                    false, ah[i].v, false, bh[j].v, (short)0, acc[i][j], false, false);
                acc[i][j] = __builtin_amdgcn_wmma_f32_16x16x32_bf16(
                    false, ah[i].v, false, bl[j].v, (short)0, acc[i][j], false, false);
                acc[i][j] = __builtin_amdgcn_wmma_f32_16x16x32_bf16(
                    false, al[i].v, false, bh[j].v, (short)0, acc[i][j], false, false);
            }
    }

    // C/D layout: VGPR v, lanes 0-15 -> M=v, N=lane; lanes 16-31 -> M=v+8, N=lane-16
    #pragma unroll
    for (int i = 0; i < 2; ++i)
        #pragma unroll
        for (int j = 0; j < 2; ++j) {
            int n = n0 + j * 16 + l15;
            int rbase = m0 + i * 16 + lhi * 8;
            #pragma unroll
            for (int v = 0; v < 8; ++v)
                gx[(size_t)(rbase + v) * GTOT + n] = acc[i][j][v];
        }
}

// ---------------------------------------------------------------------------
// Kernel 4: sequential LSTM, one block per model, 512 threads = one gate each.
// mask_h is folded into the register-pinned W_hh row (column scaling), so the
// LDS broadcast buffer holds raw h. Dot product uses float2 chunks to let the
// backend emit v_pk_fma_f32 (2 f32 FLOPs/lane/instr), 4 independent chains.
// gx row for t+1 is prefetched into a register during step t's FMA chain.
// ---------------------------------------------------------------------------
__device__ __forceinline__ float sigf(float x) {
    float e = __expf(-x);
    return __builtin_amdgcn_rcpf(1.f + e);
}
__device__ __forceinline__ float tanh_fast(float x) {
    float e = __expf(-2.f * x);
    return (1.f - e) * __builtin_amdgcn_rcpf(1.f + e);
}

__global__ __launch_bounds__(512, 1) void
lstm_kernel(const float* __restrict__ gx, const float* __restrict__ W_hh,
            const float* __restrict__ b_ih, const float* __restrict__ b_hh,
            const float* __restrict__ mask_h, float* __restrict__ Y) {
    const int r = blockIdx.x;            // model
    const int g = threadIdx.x;           // gate 0..511

    __shared__ float4 hbuf4[HID / 4];    // h broadcast buffer (mask folded into w)
    __shared__ float  gbuf[G4];          // raw gate preactivations
    float* hbuf = (float*)hbuf4;

    // pin this gate's W_hh row in registers, with mask_h folded in per column
    v2f w2[HID / 2];
    {
        const float4* wp = (const float4*)(W_hh + ((size_t)r * G4 + g) * HID);
        const float4* mp = (const float4*)(mask_h + r * HID);
        #pragma unroll
        for (int i = 0; i < HID / 4; ++i) {
            float4 v = wp[i];
            float4 m = mp[i];
            w2[2 * i + 0] = (v2f){v.x * m.x, v.y * m.y};
            w2[2 * i + 1] = (v2f){v.z * m.z, v.w * m.w};
        }
    }
    const float bsum = b_ih[r * G4 + g] + b_hh[r * G4 + g];
    float c = 0.f;

    if (g < HID) hbuf[g] = 0.f;
    __syncthreads();

    const float* gxp = gx + r * G4 + g;
    float gnext = gxp[0];

    for (int t = 0; t < T_LEN; ++t) {
        float gcur = gnext;
        int tn = (t + 1 < T_LEN) ? (t + 1) : t;
        gnext = gxp[(size_t)tn * GTOT];                 // prefetch next step

        v2f a0 = (v2f)(0.f), a1 = (v2f)(0.f), a2 = (v2f)(0.f), a3 = (v2f)(0.f);
        #pragma unroll
        for (int i = 0; i < HID / 8; ++i) {             // 16 iters x 8 elems
            float4 h0 = hbuf4[2 * i + 0];               // LDS broadcast
            float4 h1 = hbuf4[2 * i + 1];
            a0 = __builtin_elementwise_fma(w2[4 * i + 0], (v2f){h0.x, h0.y}, a0);
            a1 = __builtin_elementwise_fma(w2[4 * i + 1], (v2f){h0.z, h0.w}, a1);
            a2 = __builtin_elementwise_fma(w2[4 * i + 2], (v2f){h1.x, h1.y}, a2);
            a3 = __builtin_elementwise_fma(w2[4 * i + 3], (v2f){h1.z, h1.w}, a3);
        }
        v2f s01 = a0 + a1;
        v2f s23 = a2 + a3;
        v2f s = s01 + s23;
        gbuf[g] = gcur + bsum + (s.x + s.y);
        __syncthreads();

        if (g < HID) {
            float iv = sigf(gbuf[g]);
            float fv = sigf(gbuf[HID + g]);
            float gv = tanh_fast(gbuf[2 * HID + g]);
            float ov = sigf(gbuf[3 * HID + g]);
            c = fv * c + iv * gv;
            float h = ov * tanh_fast(c);
            Y[(size_t)t * (NM * HID) + r * HID + g] = h;          // fire-and-forget
            hbuf[g] = h;
        }
        __syncthreads();
    }
}

// ---------------------------------------------------------------------------
// Kernel 5: heads. out[t,s] = 0.2 * sum_r (W_s[r] . Y[t,r] + b_s[r])
// ---------------------------------------------------------------------------
__global__ void heads_kernel(const float* __restrict__ Y,
                             const float* __restrict__ W1, const float* __restrict__ b1,
                             const float* __restrict__ W2, const float* __restrict__ b2,
                             const float* __restrict__ W3, const float* __restrict__ b3,
                             const float* __restrict__ W4, const float* __restrict__ b4,
                             float* __restrict__ out) {
    int idx = blockIdx.x * blockDim.x + threadIdx.x;
    if (idx >= T_LEN * 15) return;
    int t = idx / 15;
    int s = idx - t * 15;

    const float4* y4 = (const float4*)(Y + (size_t)t * (NM * HID));
    float acc = 0.f, bs = 0.f;
    for (int r = 0; r < NM; ++r) {
        const float* wrow;
        if (s < 11)      { wrow = W1 + (size_t)(r * 11 + s) * HID;        bs += b1[r * 11 + s]; }
        else if (s < 13) { wrow = W2 + (size_t)(r * 2 + (s - 11)) * HID;  bs += b2[r * 2 + (s - 11)]; }
        else if (s == 13){ wrow = W3 + (size_t)r * HID;                   bs += b3[r]; }
        else             { wrow = W4 + (size_t)r * HID;                   bs += b4[r]; }
        const float4* w4 = (const float4*)wrow;
        float a = 0.f;
        #pragma unroll 8
        for (int i = 0; i < HID / 4; ++i) {
            float4 yv = y4[r * (HID / 4) + i];
            float4 wv = w4[i];
            a += yv.x * wv.x + yv.y * wv.y + yv.z * wv.z + yv.w * wv.w;
        }
        acc += a;
    }
    float val = 0.2f * (acc + bs);

    const int SL = T_LEN * 11, EX = T_LEN * 2;
    if (s < 11)       out[t * 11 + s] = val;
    else if (s < 13)  out[SL + t * 2 + (s - 11)] = val;
    else if (s == 13) out[SL + EX + t] = val;
    else              out[SL + EX + T_LEN + t] = val;
}

// ---------------------------------------------------------------------------
extern "C" void kernel_launch(void* const* d_in, const int* in_sizes, int n_in,
                              void* d_out, int out_size, void* d_ws, size_t ws_size,
                              hipStream_t stream) {
    const float* X      = (const float*)d_in[0];
    const float* W_ih   = (const float*)d_in[1];
    const float* W_hh   = (const float*)d_in[2];
    const float* b_ih   = (const float*)d_in[3];
    const float* b_hh   = (const float*)d_in[4];
    const float* W1     = (const float*)d_in[5];
    const float* b1     = (const float*)d_in[6];
    const float* W2     = (const float*)d_in[7];
    const float* b2     = (const float*)d_in[8];
    const float* W3     = (const float*)d_in[9];
    const float* b3     = (const float*)d_in[10];
    const float* W4     = (const float*)d_in[11];
    const float* b4     = (const float*)d_in[12];
    const float* mask_x = (const float*)d_in[13];
    const float* mask_h = (const float*)d_in[14];
    float* out = (float*)d_out;

    char* ws = (char*)d_ws;
    size_t o = 0;
    float* gx = (float*)(ws + o); o += (size_t)T_LEN * GTOT * sizeof(float);
    bf16*  Xh = (bf16*)(ws + o);  o += (size_t)T_LEN * FEAT * sizeof(bf16);
    bf16*  Xl = (bf16*)(ws + o);  o += (size_t)T_LEN * FEAT * sizeof(bf16);
    bf16*  Wh = (bf16*)(ws + o);  o += (size_t)GTOT * FEAT * sizeof(bf16);
    bf16*  Wl = (bf16*)(ws + o);  o += (size_t)GTOT * FEAT * sizeof(bf16);
    float* Yb = (float*)(ws + o); o += (size_t)T_LEN * NM * HID * sizeof(float);
    (void)ws_size; (void)in_sizes; (void)n_in; (void)out_size;

    {   // split X
        int n = T_LEN * FEAT;
        split_x_kernel<<<(n + 255) / 256, 256, 0, stream>>>(X, Xh, Xl);
    }
    {   // mask + split W
        int n = GTOT * FEAT;
        split_w_kernel<<<(n + 255) / 256, 256, 0, stream>>>(W_ih, mask_x, Wh, Wl);
    }
    {   // gx GEMM: grid (N/128, M/64)
        dim3 grid(GTOT / 128, T_LEN / 64);
        gemm_gx_kernel<<<grid, 256, 0, stream>>>(Xh, Xl, Wh, Wl, gx);
    }
    {   // sequential LSTM, one block per model
        lstm_kernel<<<NM, 512, 0, stream>>>(gx, W_hh, b_ih, b_hh, mask_h, Yb);
    }
    {   // heads
        int n = T_LEN * 15;
        heads_kernel<<<(n + 255) / 256, 256, 0, stream>>>(Yb, W1, b1, W2, b2, W3, b3, W4, b4, out);
    }
}
